// MoE_67242007986668
// MI455X (gfx1250) — matrix operand order repaired and checked by reading, not compiled, not run on hardware
//
#include <hip/hip_runtime.h>
#include <hip/hip_bf16.h>
#include <cstdint>
#include <cstddef>

// MoE: N=4096 tokens, D=512, E=8 (+1 shared), H=256, top-2.
// Dense 9-expert formulation; f16 WMMA (v_wmma_f32_16x16x32_f16) with fp32
// accumulation. Weight panels staged to LDS two ways:
//   - gate/up kernel: global_load_async_to_lds_b128 (ASYNCcnt), double buffer
//   - down kernel:    TDM tensor_load_to_lds (TENSORcnt), double buffer

typedef __attribute__((ext_vector_type(16))) _Float16 v16h;
typedef __attribute__((ext_vector_type(8)))  _Float16 v8h;
typedef __attribute__((ext_vector_type(8)))  float    v8f;
typedef __attribute__((ext_vector_type(4)))  unsigned v4u;
typedef __attribute__((ext_vector_type(8)))  int      v8i;
typedef __attribute__((ext_vector_type(4)))  int      v4i;

#define NTOK   4096
#define DMODEL 512
#define NEXP   8
#define NEXP1  9     // 8 routed + 1 shared
#define HDIM   256

#define BM 128       // tokens per block (8 waves x 16 rows)
#define BH 64        // gate/up output cols per block
#define BD 64        // down-proj output cols per block
#define KC 64        // K chunk staged in LDS per step

// ---------------------------------------------------------------------------
// CDNA5 async copy: global -> LDS, 16B per lane, tracked by ASYNCcnt
// ---------------------------------------------------------------------------
__device__ __forceinline__ void async_ld_b128(void* lds_dst, const void* gsrc) {
  unsigned off = (unsigned)(size_t)lds_dst;   // low 32 bits = LDS byte offset
  asm volatile("global_load_async_to_lds_b128 %0, %1, off"
               :: "v"(off), "v"(gsrc) : "memory");
}
__device__ __forceinline__ void wait_async0() {
  asm volatile("s_wait_asynccnt 0" ::: "memory");
}

// ---------------------------------------------------------------------------
// CDNA5 Tensor Data Mover: 2D tile (f16) global -> LDS, tracked by TENSORcnt
// D# built per CDNA5 ISA bit layout; groups 2/3 zero (2D tensor).
// This toolchain exposes the 6-arg builtin: (v4u, v8i, v4i, v4i, v8i, cpol).
// ---------------------------------------------------------------------------
__device__ __forceinline__ void tdm_load_2d_f16(unsigned lds_off, const void* gsrc,
                                                unsigned tensor_d0, unsigned tensor_d1,
                                                unsigned tile_d0, unsigned tile_d1,
                                                unsigned stride0) {
  size_t ga = (size_t)gsrc;
  v4u g0;
  g0[0] = 1u;                                        // count=1 (valid descriptor)
  g0[1] = lds_off;                                   // lds_addr (bytes)
  g0[2] = (unsigned)ga;                              // global_addr[31:0]
  g0[3] = (unsigned)(ga >> 32) | 0x80000000u;        // addr[56:32] | type=2<<30
  v8i g1;
  g1[0] = (int)(1u << 16);                           // wg_mask=0, data_size=1 (2B)
  g1[1] = (int)((tensor_d0 & 0xFFFFu) << 16);        // tensor_dim0[15:0] @63:48
  g1[2] = (int)((tensor_d0 >> 16) | ((tensor_d1 & 0xFFFFu) << 16));
  g1[3] = (int)((tensor_d1 >> 16) | ((tile_d0 & 0xFFFFu) << 16));
  g1[4] = (int)(tile_d1 & 0xFFFFu);                  // tile_dim1; tile_dim2=0
  g1[5] = (int)stride0;                              // tensor_dim0_stride[31:0]
  g1[6] = 0;                                         // stride0[47:32], stride1 lo
  g1[7] = 0;
  v4i g2 = {0, 0, 0, 0};
  v4i g3 = {0, 0, 0, 0};
  v8i g4 = {0, 0, 0, 0, 0, 0, 0, 0};                 // extra words (unused, 2D)
  __builtin_amdgcn_tensor_load_to_lds(g0, g1, g2, g3, g4, 0);
}
__device__ __forceinline__ void wait_tensor0() {
  __builtin_amdgcn_s_wait_tensorcnt(0);
}

// ---------------------------------------------------------------------------
// WMMA fragment loaders (wave32, 16x16x32 f16, contiguous k = 16*half + s in A and B)
// ---------------------------------------------------------------------------
// A (16x32) from row-major global: lane L -> row L&15; K-halves by L>=16.
__device__ __forceinline__ v16h load_a_frag(const _Float16* __restrict__ A,
                                            int lda, int m0, int k0) {
  const int lane = threadIdx.x & 31;
  const int r    = lane & 15;
  const int half = lane >> 4;
  const _Float16* p = A + (size_t)(m0 + r) * lda + k0;
  v8h lo = *(const v8h*)(p + half * 16);
  v8h hi = *(const v8h*)(p + half * 16 + 8);
  v16h a;
#pragma unroll
  for (int i = 0; i < 8; ++i) { a[i] = lo[i]; a[i + 8] = hi[i]; }
  return a;
}

// B (32x16) from an LDS tile laid out [16 cols][KC] (K contiguous).
__device__ __forceinline__ v16h load_b_lds(const _Float16* Bt, int k0) {
  const int lane = threadIdx.x & 31;
  const int n  = lane & 15;
  const int kb = k0 + (lane >> 4) * 16;
  return *(const v16h*)(Bt + n * KC + kb);   // 32B -> 2x ds_load_b128
}

__device__ __forceinline__ v8f wmma16(v16h a, v16h b, v8f c) {
  return __builtin_amdgcn_wmma_f32_16x16x32_f16(false, a, false, b,
                                                (short)0, c, false, false);
}

// ---------------------------------------------------------------------------
// Prep kernels
// ---------------------------------------------------------------------------
__global__ void cvt_x_kernel(const float* __restrict__ x,
                             _Float16* __restrict__ xh, int n) {
  int i = blockIdx.x * blockDim.x + threadIdx.x;
  if (i < n) xh[i] = (_Float16)x[i];
}

// WT[e][c][r] = f16( e<8 ? W[e][r][c] : Wsh[r][c] )  with W: [8][rows][cols]
__global__ void transpose_w_kernel(const float* __restrict__ W,
                                   const float* __restrict__ Wsh,
                                   _Float16* __restrict__ WT,
                                   int rows, int cols) {
  size_t per = (size_t)rows * cols;
  size_t i = (size_t)blockIdx.x * blockDim.x + threadIdx.x;
  if (i >= (size_t)NEXP1 * per) return;
  int e = (int)(i / per);
  size_t r2 = i % per;
  int c = (int)(r2 / rows);
  int r = (int)(r2 % rows);
  float v = (e < NEXP) ? W[(size_t)e * per + (size_t)r * cols + c]
                       : Wsh[(size_t)r * cols + c];
  WT[i] = (_Float16)v;
}

// ---------------------------------------------------------------------------
// Router: logits -> softmax -> top-2 -> dense combine [N][9] (col 8 = shared)
// ---------------------------------------------------------------------------
__global__ void router_kernel(const float* __restrict__ x,
                              const float* __restrict__ router,
                              float* __restrict__ combine) {
  int t = blockIdx.x * blockDim.x + threadIdx.x;
  if (t >= NTOK) return;
  const float* xr = x + (size_t)t * DMODEL;
  float logit[NEXP];
#pragma unroll
  for (int e = 0; e < NEXP; ++e) logit[e] = 0.0f;
  for (int k = 0; k < DMODEL; ++k) {
    float xv = xr[k];
    const float* rr = router + (size_t)k * NEXP;
#pragma unroll
    for (int e = 0; e < NEXP; ++e) logit[e] = fmaf(xv, rr[e], logit[e]);
  }
  float m = logit[0];
#pragma unroll
  for (int e = 1; e < NEXP; ++e) m = fmaxf(m, logit[e]);
  float p[NEXP]; float s = 0.0f;
#pragma unroll
  for (int e = 0; e < NEXP; ++e) { p[e] = __expf(logit[e] - m); s += p[e]; }
  float inv = 1.0f / s;
#pragma unroll
  for (int e = 0; e < NEXP; ++e) p[e] *= inv;
  int i1 = 0;
#pragma unroll
  for (int e = 1; e < NEXP; ++e) if (p[e] > p[i1]) i1 = e;
  int i2 = (i1 == 0) ? 1 : 0;
#pragma unroll
  for (int e = 0; e < NEXP; ++e) if (e != i1 && p[e] > p[i2]) i2 = e;
  float* crow = combine + (size_t)t * NEXP1;
#pragma unroll
  for (int e = 0; e < NEXP; ++e) crow[e] = 0.0f;
  crow[i1] = p[i1];
  crow[i2] = p[i2];
  crow[NEXP] = 1.0f;
}

// ---------------------------------------------------------------------------
// Gate/Up + SwiGLU + combine fold: Hs[e][n][h] = comb[n][e]*silu(g)*u  (f16)
// Block: 256 thr (8 waves). Tile: 128 tokens x 64 H-cols. B panels (gate+up)
// double-buffered in LDS via per-lane async copies (ASYNCcnt).
// ---------------------------------------------------------------------------
__global__ __launch_bounds__(256)
void moe_gateup_kernel(const _Float16* __restrict__ xh,
                       const _Float16* __restrict__ WgT,
                       const _Float16* __restrict__ WuT,
                       const float* __restrict__ combine,
                       _Float16* __restrict__ Hs) {
  __shared__ __align__(32) _Float16 Bsm[2][2][BH][KC];   // 32 KB
  const int h0   = blockIdx.x * BH;
  const int m0   = blockIdx.y * BM;
  const int e    = blockIdx.z;
  const int tid  = threadIdx.x;
  const int wave = tid >> 5;
  const int lane = tid & 31;
  const int mrow = m0 + wave * 16;

  const _Float16* wg = WgT + (size_t)e * HDIM * DMODEL;
  const _Float16* wu = WuT + (size_t)e * HDIM * DMODEL;

  auto stage = [&](int kc, int buf) {
    // 2 matrices * 64 rows * 8 x16B segments = 1024 chunks; 4 per thread
#pragma unroll
    for (int j = 0; j < 4; ++j) {
      int c   = tid + 256 * j;
      int mat = c >> 9;
      int row = (c >> 3) & 63;
      int seg = c & 7;
      const _Float16* src =
          (mat ? wu : wg) + (size_t)(h0 + row) * DMODEL + kc + seg * 8;
      async_ld_b128(&Bsm[buf][mat][row][seg * 8], src);
    }
  };

  v8f g[4] = {}; v8f u[4] = {};

  stage(0, 0);
  for (int kc = 0; kc < DMODEL; kc += KC) {
    const int buf = (kc / KC) & 1;
    wait_async0();
    __syncthreads();
    if (kc + KC < DMODEL) stage(kc + KC, buf ^ 1);
#pragma unroll
    for (int kk = 0; kk < KC; kk += 32) {
      v16h a = load_a_frag(xh, DMODEL, mrow, kc + kk);
#pragma unroll
      for (int ht = 0; ht < 4; ++ht) {
        v16h bg = load_b_lds(&Bsm[buf][0][ht * 16][0], kk);
        v16h bu = load_b_lds(&Bsm[buf][1][ht * 16][0], kk);
        g[ht] = wmma16(a, bg, g[ht]);
        u[ht] = wmma16(a, bu, u[ht]);
      }
    }
  }

  // epilogue: silu(g)*u * combine[row][e], f16 store
  const int col   = lane & 15;
  const int rbase = mrow + (lane >> 4) * 8;
  float cw[8];
#pragma unroll
  for (int v = 0; v < 8; ++v)
    cw[v] = combine[(size_t)(rbase + v) * NEXP1 + e];
  _Float16* outp = Hs + (size_t)e * NTOK * HDIM;
#pragma unroll
  for (int ht = 0; ht < 4; ++ht) {
#pragma unroll
    for (int v = 0; v < 8; ++v) {
      float gv = g[ht][v];
      float hv = (gv / (1.0f + __expf(-gv))) * u[ht][v] * cw[v];
      outp[(size_t)(rbase + v) * HDIM + h0 + ht * 16 + col] = (_Float16)hv;
    }
  }
}

// ---------------------------------------------------------------------------
// Down-proj accumulate: out[n][d] = sum_e Hs[e][n][:] @ Wd_e[:][d]   (fp32)
// Block: 256 thr (8 waves). Tile: 128 tokens x 64 D-cols. 36 K-chunks
// (9 experts x 4). B staged by the Tensor Data Mover: wave 0 issues one
// tensor_load_to_lds (64x64 f16 tile, row stride HDIM) per chunk, waits on
// TENSORcnt, then the workgroup barrier publishes the buffer to all 8 waves.
// ---------------------------------------------------------------------------
__global__ __launch_bounds__(256)
void moe_down_kernel(const _Float16* __restrict__ Hs,
                     const _Float16* __restrict__ WdT,
                     float* __restrict__ out) {
  __shared__ __align__(32) _Float16 Bsm[2][BD][KC];      // 16 KB
  const int d0   = blockIdx.x * BD;
  const int m0   = blockIdx.y * BM;
  const int tid  = threadIdx.x;
  const int wave = tid >> 5;
  const int lane = tid & 31;
  const int mrow = m0 + wave * 16;

  const int NCH = NEXP1 * (HDIM / KC);   // 36 chunks

  auto tdm_stage = [&](int ch, int buf) {
    const int ce = ch >> 2;
    const int ck = (ch & 3) * KC;
    const _Float16* src =
        WdT + (size_t)ce * DMODEL * HDIM + (size_t)d0 * HDIM + ck;
    tdm_load_2d_f16((unsigned)(size_t)&Bsm[buf][0][0], src,
                    /*tensor_d0=*/HDIM, /*tensor_d1=*/BD,
                    /*tile_d0=*/KC, /*tile_d1=*/BD,
                    /*stride0=*/HDIM);
  };

  v8f acc[4] = {};

  if (wave == 0) tdm_stage(0, 0);
  for (int ch = 0; ch < NCH; ++ch) {
    const int buf = ch & 1;
    const int e   = ch >> 2;
    const int kc  = (ch & 3) * KC;
    if (wave == 0) wait_tensor0();
    __syncthreads();                       // buffer `buf` visible to all waves
    if (wave == 0 && ch + 1 < NCH) tdm_stage(ch + 1, buf ^ 1);
    const _Float16* A = Hs + (size_t)e * NTOK * HDIM;
#pragma unroll
    for (int kk = 0; kk < KC; kk += 32) {
      v16h a = load_a_frag(A, HDIM, mrow, kc + kk);
#pragma unroll
      for (int dt = 0; dt < 4; ++dt) {
        v16h b = load_b_lds(&Bsm[buf][dt * 16][0], kk);
        acc[dt] = wmma16(a, b, acc[dt]);
      }
    }
  }

  const int col   = lane & 15;
  const int rbase = mrow + (lane >> 4) * 8;
#pragma unroll
  for (int dt = 0; dt < 4; ++dt)
#pragma unroll
    for (int v = 0; v < 8; ++v)
      out[(size_t)(rbase + v) * DMODEL + d0 + dt * 16 + col] = acc[dt][v];
}

// ---------------------------------------------------------------------------
extern "C" void kernel_launch(void* const* d_in, const int* in_sizes, int n_in,
                              void* d_out, int out_size, void* d_ws, size_t ws_size,
                              hipStream_t stream) {
  const float* x   = (const float*)d_in[0];   // [4,1024,512]
  const float* rt  = (const float*)d_in[1];   // [512,8]
  const float* sg  = (const float*)d_in[2];   // [512,256]
  const float* su  = (const float*)d_in[3];   // [512,256]
  const float* sd  = (const float*)d_in[4];   // [256,512]
  const float* Wg  = (const float*)d_in[5];   // [8,512,256]
  const float* Wu  = (const float*)d_in[6];   // [8,512,256]
  const float* Wd  = (const float*)d_in[7];   // [8,256,512]
  float* out = (float*)d_out;                 // [4,1024,512] fp32

  char* ws = (char*)d_ws;
  size_t off = 0;
  auto take = [&](size_t bytes) -> void* {
    void* p = ws + off;
    off = (off + bytes + 255) & ~(size_t)255;
    return p;
  };
  _Float16* xh   = (_Float16*)take((size_t)NTOK * DMODEL * sizeof(_Float16));
  _Float16* WgT  = (_Float16*)take((size_t)NEXP1 * HDIM * DMODEL * sizeof(_Float16));
  _Float16* WuT  = (_Float16*)take((size_t)NEXP1 * HDIM * DMODEL * sizeof(_Float16));
  _Float16* WdT  = (_Float16*)take((size_t)NEXP1 * DMODEL * HDIM * sizeof(_Float16));
  _Float16* Hs   = (_Float16*)take((size_t)NEXP1 * NTOK * HDIM * sizeof(_Float16));
  float*    comb = (float*)take((size_t)NTOK * NEXP1 * sizeof(float));
  (void)ws_size; (void)in_sizes; (void)n_in; (void)out_size;

  {
    int n = NTOK * DMODEL;
    cvt_x_kernel<<<dim3((n + 255) / 256), dim3(256), 0, stream>>>(x, xh, n);
  }
  {
    size_t tot = (size_t)NEXP1 * DMODEL * HDIM;
    dim3 g((unsigned)((tot + 255) / 256));
    transpose_w_kernel<<<g, 256, 0, stream>>>(Wg, sg, WgT, DMODEL, HDIM);
    transpose_w_kernel<<<g, 256, 0, stream>>>(Wu, su, WuT, DMODEL, HDIM);
    transpose_w_kernel<<<g, 256, 0, stream>>>(Wd, sd, WdT, HDIM, DMODEL);
  }

  router_kernel<<<dim3(NTOK / 256), dim3(256), 0, stream>>>(x, rt, comb);

  {
    dim3 grid(HDIM / BH, NTOK / BM, NEXP1);   // (4, 32, 9)
    moe_gateup_kernel<<<grid, dim3(256), 0, stream>>>(xh, WgT, WuT, comb, Hs);
  }
  {
    dim3 grid(DMODEL / BD, NTOK / BM);        // (8, 32)
    moe_down_kernel<<<grid, dim3(256), 0, stream>>>(Hs, WdT, out);
  }
}
